// EfficientDGCNNBackbone_28518582845516
// MI455X (gfx1250) — matrix operand order repaired and checked by reading, not compiled, not run on hardware
//
#include <hip/hip_runtime.h>
#include <hip/hip_bf16.h>

typedef __attribute__((ext_vector_type(16))) _Float16 v16h;
typedef __attribute__((ext_vector_type(8)))  float    v8f;

#define DEVI static __device__ __forceinline__

DEVI int kbA(int r, int h) { return (r < 4) ? (2*r + 8*h) : (16 + 2*(r-4) + 8*h); }
DEVI float lrelu(float v) { return v >= 0.f ? v : 0.2f * v; }

// Fragment loaders matching the CDNA5 16x16x32 f16 WMMA VGPR layouts.
DEVI void load_fragA(const _Float16* __restrict__ row, int kc, int lh, v16h& a) {
  #pragma unroll
  for (int r = 0; r < 8; ++r) {
    int ka = kc * 32 + kbA(r, lh);
    a[2 * r] = row[ka]; a[2 * r + 1] = row[ka + 1];
  }
}
DEVI void load_fragB(const _Float16* __restrict__ row, int kc, int lh, v16h& b) {
  #pragma unroll
  for (int r = 0; r < 8; ++r) {
    int kb = kc * 32 + kbA(r, lh);
    b[2 * r] = row[kb]; b[2 * r + 1] = row[kb + 1];
  }
}
DEVI v8f wmma16(v16h a, v16h b, v8f c) {
  return __builtin_amdgcn_wmma_f32_16x16x32_f16(false, a, false, b, (short)0, c,
                                                false, false);
}

// ---------------------------------------------------------------- small utils
__global__ void k_padcvt(const float* __restrict__ src, _Float16* __restrict__ dst,
                         int rows, int kin, int kout) {
  int i = blockIdx.x * blockDim.x + threadIdx.x;
  if (i >= rows * kout) return;
  int r = i / kout, k = i - r * kout;
  dst[i] = (k < kin) ? (_Float16)src[r * kin + k] : (_Float16)0.f;
}

__global__ void k_bnfold(const float* __restrict__ g, const float* __restrict__ b,
                         const float* __restrict__ m, const float* __restrict__ v,
                         float* __restrict__ scale, float* __restrict__ shift, int n) {
  int i = blockIdx.x * blockDim.x + threadIdx.x;
  if (i >= n) return;
  float s = g[i] * rsqrtf(v[i] + 1e-5f);
  scale[i] = s;
  shift[i] = b[i] - m[i] * s;
}

__global__ void k_headcols(const float* s0, const float* h0, const float* b0,
                           const float* s1, const float* h1, const float* b1,
                           const float* s2, const float* h2, const float* b2,
                           const float* s3, const float* h3, const float* b3,
                           float* cs, float* ch, float* cb) {
  int o = blockIdx.x * blockDim.x + threadIdx.x;
  if (o >= 768) return;
  const float *s, *h, *b; int c;
  if (o < 256)      { s = s0; h = h0; b = b0; c = o; }
  else if (o < 512) { s = s1; h = h1; b = b1; c = o - 256; }
  else if (o < 640) { s = s2; h = h2; b = b2; c = o - 512; }
  else              { s = s3; h = h3; b = b3; c = o - 640; }
  cs[o] = s[c]; ch[o] = h[c]; cb[o] = b[c];
}

__global__ void k_padx0(const float* __restrict__ x, _Float16* __restrict__ x0) {
  // x: (4,3,4096) f32 -> x0: (4*4096, 32) f16 zero-padded
  int i = blockIdx.x * blockDim.x + threadIdx.x;
  if (i >= 4 * 4096 * 32) return;
  int c = i & 31, bn = i >> 5, b = bn >> 12, n = bn & 4095;
  x0[i] = (c < 3) ? (_Float16)x[(b * 3 + c) * 4096 + n] : (_Float16)0.f;
}

// -------------------------------------------------- fused Gram + top-K kernel
// Block: 16 query rows, 256 threads (8 waves). Each wave computes one 16x16
// inner-product tile per 128-column chunk; all B fragments for the tile are
// batch-loaded first (loads in flight), then WMMAs run back-to-back.
template<int CP>
__global__ __launch_bounds__(256)
void k_dist_topk(const _Float16* __restrict__ feat, int stride, int colofs,
                 int* __restrict__ idxbuf) {
  extern __shared__ char smem[];
  _Float16* xq  = (_Float16*)smem;                    // 16*CP
  float* xxq    = (float*)(xq + 16 * CP);             // 16
  float* dtile  = xxq + 16;                           // 8*16*16
  float* mval   = dtile + 8 * 16 * 16;                // 16*16*10
  int*   midx   = (int*)(mval + 16 * 16 * 10);        // 16*16*10

  const int tid = threadIdx.x;
  const int w = tid >> 5, l = tid & 31, lm = l & 15, lh = l >> 4;
  const int bx = blockIdx.x;
  const int b = bx >> 8, i0 = (bx & 255) << 4;
  const size_t bN = (size_t)b * 4096;

  for (int e = tid; e < 16 * CP; e += 256) {
    int rr = e / CP, c = e - rr * CP;
    xq[e] = feat[(bN + i0 + rr) * (size_t)stride + colofs + c];
  }
  __syncthreads();
  if (tid < 16) {
    float s = 0.f;
    for (int c = 0; c < CP; ++c) { float a = (float)xq[tid * CP + c]; s += a * a; }
    xxq[tid] = s;
  }
  __syncthreads();

  constexpr int NK = CP / 32;
  v16h afrag[NK];
  #pragma unroll
  for (int kc = 0; kc < NK; ++kc) load_fragA(xq + lm * CP, kc, lh, afrag[kc]);

  float kv[10]; int ki[10];
  #pragma unroll
  for (int q = 0; q < 10; ++q) { kv[q] = -3.0e38f; ki[q] = 0; }
  const int rt = tid >> 4, ct = tid & 15;

  for (int j0 = 0; j0 < 4096; j0 += 128) {
    const int jc = j0 + 16 * w;
    const _Float16* rowp = feat + (bN + jc + lm) * (size_t)stride + colofs;
    // batch-load all B fragments for this tile (keeps loads in flight)
    v16h bfr[NK];
    #pragma unroll
    for (int kc = 0; kc < NK; ++kc) load_fragB(rowp, kc, lh, bfr[kc]);
    float xj2 = 0.f;
    #pragma unroll
    for (int kc = 0; kc < NK; ++kc)
      #pragma unroll
      for (int e = 0; e < 16; ++e) { float f = (float)bfr[kc][e]; xj2 += f * f; }
    v8f acc = {};
    #pragma unroll
    for (int kc = 0; kc < NK; ++kc) acc = wmma16(afrag[kc], bfr[kc], acc);

    xj2 += __shfl_xor(xj2, 16, 32);   // combine k-halves for this column
    #pragma unroll
    for (int r = 0; r < 8; ++r) {
      int row = r + 8 * lh;
      dtile[(w * 16 + row) * 16 + lm] = 2.f * acc[r] - xxq[row] - xj2;
    }
    __syncthreads();
    #pragma unroll
    for (int ww = 0; ww < 8; ++ww) {
      float v = dtile[(ww * 16 + rt) * 16 + ct];
      int j = j0 + 16 * ww + ct;
      if (v > kv[9]) {
        int pos = 0;
        #pragma unroll
        for (int q = 0; q < 9; ++q) pos += (kv[q] >= v) ? 1 : 0;
        #pragma unroll
        for (int q = 9; q >= 1; --q) if (q > pos) { kv[q] = kv[q-1]; ki[q] = ki[q-1]; }
        kv[pos] = v; ki[pos] = j;
      }
    }
    __syncthreads();
  }

  #pragma unroll
  for (int q = 0; q < 10; ++q) {
    mval[(rt * 16 + ct) * 10 + q] = kv[q];
    midx[(rt * 16 + ct) * 10 + q] = ki[q];
  }
  __syncthreads();
  if (tid < 16) {
    float fv[10]; int fi[10];
    #pragma unroll
    for (int q = 0; q < 10; ++q) { fv[q] = -3.0e38f; fi[q] = 0; }
    for (int s = 0; s < 16; ++s) {
      for (int q = 0; q < 10; ++q) {
        float v = mval[(tid * 16 + s) * 10 + q];
        int j = midx[(tid * 16 + s) * 10 + q];
        if (v > fv[9]) {
          int pos = 0;
          #pragma unroll
          for (int qq = 0; qq < 9; ++qq) pos += (fv[qq] >= v) ? 1 : 0;
          #pragma unroll
          for (int qq = 9; qq >= 1; --qq) if (qq > pos) { fv[qq] = fv[qq-1]; fi[qq] = fi[qq-1]; }
          fv[pos] = v; fi[pos] = j;
        }
      }
    }
    #pragma unroll
    for (int q = 0; q < 10; ++q) idxbuf[(bN + i0 + tid) * 16 + q] = fi[q];
  }
}

// ------------------------------------------------------- edge-conv (gather+GEMM)
// Block: 16 points -> 160 edge rows in LDS. Each wave-task computes a 16x32
// output slab: one shared A (edge) fragment feeds two B (weight) tiles with
// independent accumulator chains; then max-over-k -> BN -> LeakyReLU.
template<int CI, int KP2, int CO>
__global__ __launch_bounds__(256)
void k_edgeconv(const _Float16* __restrict__ feat, int stride, int colofs,
                const int* __restrict__ idxbuf,
                const _Float16* __restrict__ W,          // CO x KP2 (f16)
                const float* __restrict__ scale, const float* __restrict__ shift,
                _Float16* __restrict__ outcat, int outcol) {
  extern __shared__ char smem[];
  _Float16* E   = (_Float16*)smem;                 // 160*KP2
  _Float16* ctr = E + 160 * KP2;                   // 16*CI
  float* hbuf   = (float*)(ctr + 16 * CI);         // 160*64
  int* jidx     = (int*)(hbuf + 160 * 64);         // 160

  const int tid = threadIdx.x;
  const int w = tid >> 5, l = tid & 31, lm = l & 15, lh = l >> 4;
  const int bx = blockIdx.x;
  const int b = bx >> 8, n0 = (bx & 255) << 4;
  const size_t bN = (size_t)b * 4096;

  if (tid < 160) jidx[tid] = idxbuf[(bN + n0 + tid / 10) * 16 + (tid % 10)];
  for (int e = tid; e < 16 * CI; e += 256) {
    int p = e / CI, c = e - p * CI;
    ctr[e] = feat[(bN + n0 + p) * (size_t)stride + colofs + c];
  }
  __syncthreads();
  for (int e = tid; e < 160 * KP2; e += 256) {
    int row = e / KP2, c = e - row * KP2;
    _Float16 v = (_Float16)0.f;
    if (c < CI) {
      int j = jidx[row];
      float nb = (float)feat[(bN + j) * (size_t)stride + colofs + c];
      v = (_Float16)(nb - (float)ctr[(row / 10) * CI + c]);
    } else if (c < 2 * CI) {
      v = ctr[(row / 10) * CI + (c - CI)];
    }
    E[e] = v;
  }
  __syncthreads();

  constexpr int NK = KP2 / 32;
  for (int oc0 = 0; oc0 < CO; oc0 += 64) {
    // 20 wave-tasks: 10 M-tiles x 2 N-pairs; A fragment shared by both N-tiles.
    for (int t = w; t < 20; t += 8) {
      int mt = t >> 1, np = t & 1;
      const _Float16* erow  = E + (16 * mt + lm) * KP2;
      const _Float16* wrow0 = W + (size_t)(oc0 + 32 * np + lm) * KP2;
      const _Float16* wrow1 = wrow0 + (size_t)16 * KP2;
      v8f acc0 = {}, acc1 = {};
      #pragma unroll
      for (int kc = 0; kc < NK; ++kc) {
        v16h a, b0, b1;
        load_fragA(erow, kc, lh, a);
        load_fragB(wrow0, kc, lh, b0);
        load_fragB(wrow1, kc, lh, b1);
        acc0 = wmma16(a, b0, acc0);
        acc1 = wmma16(a, b1, acc1);
      }
      #pragma unroll
      for (int r = 0; r < 8; ++r) {
        hbuf[(16 * mt + r + 8 * lh) * 64 + 32 * np + lm]      = acc0[r];
        hbuf[(16 * mt + r + 8 * lh) * 64 + 32 * np + 16 + lm] = acc1[r];
      }
    }
    __syncthreads();
    for (int e = tid; e < 16 * 64; e += 256) {
      int p = e >> 6, c = e & 63;
      float m = -3.0e38f;
      #pragma unroll
      for (int kk = 0; kk < 10; ++kk) m = fmaxf(m, hbuf[(p * 10 + kk) * 64 + c]);
      int oc = oc0 + c;
      float val = lrelu(scale[oc] * m + shift[oc]);
      outcat[(bN + n0 + p) * 512 + outcol + oc] = (_Float16)val;
    }
    __syncthreads();
  }
}

// ------------------------------------------------------------ dense GEMMs K=512
// 64x64 block tile, 8 waves, each wave = 1 M-tile x 2 N-tiles (shared A frag,
// two independent WMMA chains).
__global__ __launch_bounds__(256)
void k_gemm_feat(const _Float16* __restrict__ A,   // 16384 x 512 (cat)
                 const _Float16* __restrict__ W,   // 512 x 512
                 const float* __restrict__ scale, const float* __restrict__ shift,
                 _Float16* __restrict__ out) {     // 16384 x 512 feat
  const int tid = threadIdx.x, w = tid >> 5, l = tid & 31, lm = l & 15, lh = l >> 4;
  const int m0 = blockIdx.x * 64, o0b = blockIdx.y * 64;
  const int mt = w & 3, np = w >> 2;            // 4 M-tiles x 2 N-pairs
  const _Float16* arow  = A + (size_t)(m0 + 16 * mt + lm) * 512;
  const _Float16* wrow0 = W + (size_t)(o0b + 32 * np + lm) * 512;
  const _Float16* wrow1 = wrow0 + (size_t)16 * 512;
  v8f acc0 = {}, acc1 = {};
  #pragma unroll
  for (int kc = 0; kc < 16; ++kc) {
    v16h a, b0, b1;
    load_fragA(arow, kc, lh, a);
    load_fragB(wrow0, kc, lh, b0);
    load_fragB(wrow1, kc, lh, b1);
    acc0 = wmma16(a, b0, acc0);
    acc1 = wmma16(a, b1, acc1);
  }
  int o0 = o0b + 32 * np + lm, o1 = o0 + 16;
  float s0 = scale[o0], sh0 = shift[o0];
  float s1 = scale[o1], sh1 = shift[o1];
  #pragma unroll
  for (int r = 0; r < 8; ++r) {
    int m = m0 + 16 * mt + r + 8 * lh;
    out[(size_t)m * 512 + o0] = (_Float16)lrelu(s0 * acc0[r] + sh0);
    out[(size_t)m * 512 + o1] = (_Float16)lrelu(s1 * acc1[r] + sh1);
  }
}

DEVI void head_addr(int o, size_t& base, int& chd, int& cl) {
  if (o < 256)      { base = 0u;        chd = 256; cl = o; }
  else if (o < 512) { base = 4194304u;  chd = 256; cl = o - 256; }
  else if (o < 640) { base = 8388608u;  chd = 128; cl = o - 512; }
  else              { base = 10485760u; chd = 128; cl = o - 640; }
}

__global__ __launch_bounds__(256)
void k_gemm_heads(const _Float16* __restrict__ A,  // 16384 x 512 feat
                  const _Float16* __restrict__ W,  // 768 x 512 fused heads
                  const float* __restrict__ cs, const float* __restrict__ ch,
                  const float* __restrict__ cb,
                  float* __restrict__ out) {
  const int tid = threadIdx.x, w = tid >> 5, l = tid & 31, lm = l & 15, lh = l >> 4;
  const int m0 = blockIdx.x * 64, o0b = blockIdx.y * 64;
  const int mt = w & 3, np = w >> 2;
  const _Float16* arow  = A + (size_t)(m0 + 16 * mt + lm) * 512;
  const _Float16* wrow0 = W + (size_t)(o0b + 32 * np + lm) * 512;
  const _Float16* wrow1 = wrow0 + (size_t)16 * 512;
  v8f acc0 = {}, acc1 = {};
  #pragma unroll
  for (int kc = 0; kc < 16; ++kc) {
    v16h a, b0, b1;
    load_fragA(arow, kc, lh, a);
    load_fragB(wrow0, kc, lh, b0);
    load_fragB(wrow1, kc, lh, b1);
    acc0 = wmma16(a, b0, acc0);
    acc1 = wmma16(a, b1, acc1);
  }
  int o0 = o0b + 32 * np + lm, o1 = o0 + 16;
  size_t base0, base1; int chd0, chd1, cl0, cl1;
  head_addr(o0, base0, chd0, cl0);
  head_addr(o1, base1, chd1, cl1);
  float s0 = cs[o0], sh0 = ch[o0], bi0 = cb[o0];
  float s1 = cs[o1], sh1 = ch[o1], bi1 = cb[o1];
  #pragma unroll
  for (int r = 0; r < 8; ++r) {
    int m = m0 + 16 * mt + r + 8 * lh;
    out[base0 + (size_t)m * chd0 + cl0] = s0 * (acc0[r] + bi0) + sh0;
    out[base1 + (size_t)m * chd1 + cl1] = s1 * (acc1[r] + bi1) + sh1;
  }
}

__global__ void k_sem(const float* __restrict__ out3, const float* __restrict__ Wsem,
                      const float* __restrict__ bsem, float* __restrict__ sem) {
  int gid = blockIdx.x * blockDim.x + threadIdx.x;
  int m = gid >> 5, o = gid & 31;
  if (m >= 16384 || o >= 20) return;
  const float* xr = out3 + (size_t)m * 128;
  const float* wr = Wsem + o * 128;
  float s = bsem[o];
  for (int c = 0; c < 128; ++c) s += xr[c] * wr[c];
  sem[(size_t)m * 20 + o] = s;
}

// --------------------------------------------------------------------- launch
extern "C" void kernel_launch(void* const* d_in, const int* in_sizes, int n_in,
                              void* d_out, int out_size, void* d_ws, size_t ws_size,
                              hipStream_t stream) {
  (void)in_sizes; (void)n_in; (void)out_size; (void)ws_size;
  // Input order: recursive insertion order of setup_inputs() dict.
  const float* x    = (const float*)d_in[0];
  const float* W1   = (const float*)d_in[1];
  const float* W2   = (const float*)d_in[6];
  const float* W3   = (const float*)d_in[11];
  const float* W4   = (const float*)d_in[16];
  const float* W5   = (const float*)d_in[21];
  const float* Wf0  = (const float*)d_in[26];
  const float* Wf1  = (const float*)d_in[32];
  const float* Wf2  = (const float*)d_in[38];
  const float* Wf3  = (const float*)d_in[44];
  const float* Wsem = (const float*)d_in[50];
  const float* bsem = (const float*)d_in[51];

  char* ws = (char*)d_ws;
  size_t off = 0;
  auto alloc = [&](size_t bytes) -> void* {
    off = (off + 255) & ~(size_t)255;
    void* p = ws + off; off += bytes; return p;
  };
  _Float16* x0h   = (_Float16*)alloc((size_t)4*4096*32*2);
  _Float16* cat16 = (_Float16*)alloc((size_t)4*4096*512*2);
  _Float16* fe16  = (_Float16*)alloc((size_t)4*4096*512*2);
  int*      idxb  = (int*)alloc((size_t)4*4096*16*4);
  _Float16* W1p   = (_Float16*)alloc(64*32*2);
  _Float16* W2h   = (_Float16*)alloc(64*128*2);
  _Float16* W3h   = (_Float16*)alloc(128*128*2);
  _Float16* W4h   = (_Float16*)alloc(256*256*2);
  _Float16* W5h   = (_Float16*)alloc(512*512*2);
  _Float16* Wcat  = (_Float16*)alloc(768*512*2);
  float* s1 = (float*)alloc(64*4);   float* h1 = (float*)alloc(64*4);
  float* s2 = (float*)alloc(64*4);   float* h2 = (float*)alloc(64*4);
  float* s3 = (float*)alloc(128*4);  float* h3 = (float*)alloc(128*4);
  float* s4 = (float*)alloc(256*4);  float* h4 = (float*)alloc(256*4);
  float* s5 = (float*)alloc(512*4);  float* h5 = (float*)alloc(512*4);
  float* os0 = (float*)alloc(256*4); float* oh0 = (float*)alloc(256*4);
  float* os1 = (float*)alloc(256*4); float* oh1 = (float*)alloc(256*4);
  float* os2 = (float*)alloc(128*4); float* oh2 = (float*)alloc(128*4);
  float* os3 = (float*)alloc(128*4); float* oh3 = (float*)alloc(128*4);
  float* cscale = (float*)alloc(768*4);
  float* cshift = (float*)alloc(768*4);
  float* cbias  = (float*)alloc(768*4);

  auto cvt = [&](const float* s, _Float16* d, int rows, int kin, int kout) {
    int tot = rows * kout;
    k_padcvt<<<(tot + 255) / 256, 256, 0, stream>>>(s, d, rows, kin, kout);
  };
  cvt(W1, W1p, 64, 6, 32);
  cvt(W2, W2h, 64, 128, 128);
  cvt(W3, W3h, 128, 128, 128);
  cvt(W4, W4h, 256, 256, 256);
  cvt(W5, W5h, 512, 512, 512);
  cvt(Wf0, Wcat,             256, 512, 512);
  cvt(Wf1, Wcat + 256 * 512, 256, 512, 512);
  cvt(Wf2, Wcat + 512 * 512, 128, 512, 512);
  cvt(Wf3, Wcat + 640 * 512, 128, 512, 512);

  auto bnf = [&](int gi, float* s, float* h, int n) {
    k_bnfold<<<(n + 255) / 256, 256, 0, stream>>>(
        (const float*)d_in[gi], (const float*)d_in[gi + 1],
        (const float*)d_in[gi + 2], (const float*)d_in[gi + 3], s, h, n);
  };
  bnf(2, s1, h1, 64);   bnf(7, s2, h2, 64);
  bnf(12, s3, h3, 128); bnf(17, s4, h4, 256);
  bnf(22, s5, h5, 512);
  bnf(28, os0, oh0, 256); bnf(34, os1, oh1, 256);
  bnf(40, os2, oh2, 128); bnf(46, os3, oh3, 128);

  k_headcols<<<3, 256, 0, stream>>>(os0, oh0, (const float*)d_in[27],
                                    os1, oh1, (const float*)d_in[33],
                                    os2, oh2, (const float*)d_in[39],
                                    os3, oh3, (const float*)d_in[45],
                                    cscale, cshift, cbias);

  k_padx0<<<(4 * 4096 * 32) / 256, 256, 0, stream>>>(x, x0h);

  auto dist_sm = [](int cp) {
    return (size_t)16 * cp * 2 + 16 * 4 + 8 * 16 * 16 * 4 + 16 * 16 * 10 * 4 * 2;
  };
  auto edge_sm = [](int ci, int kp2) {
    return (size_t)160 * kp2 * 2 + 16 * ci * 2 + 160 * 64 * 4 + 160 * 4;
  };

  // Layer 1: x0 (C=3 padded to 32) -> cat cols [0,64)
  k_dist_topk<32><<<1024, 256, dist_sm(32), stream>>>(x0h, 32, 0, idxb);
  k_edgeconv<3, 32, 64><<<1024, 256, edge_sm(3, 32), stream>>>(
      x0h, 32, 0, idxb, W1p, s1, h1, cat16, 0);
  // Layer 2: x1 cols [0,64) -> cat cols [64,128)
  k_dist_topk<64><<<1024, 256, dist_sm(64), stream>>>(cat16, 512, 0, idxb);
  k_edgeconv<64, 128, 64><<<1024, 256, edge_sm(64, 128), stream>>>(
      cat16, 512, 0, idxb, W2h, s2, h2, cat16, 64);
  // Layer 3: x2 cols [64,128) -> cat cols [128,256)
  k_dist_topk<64><<<1024, 256, dist_sm(64), stream>>>(cat16, 512, 64, idxb);
  k_edgeconv<64, 128, 128><<<1024, 256, edge_sm(64, 128), stream>>>(
      cat16, 512, 64, idxb, W3h, s3, h3, cat16, 128);
  // Layer 4: x3 cols [128,256) -> cat cols [256,512)
  k_dist_topk<128><<<1024, 256, dist_sm(128), stream>>>(cat16, 512, 128, idxb);
  k_edgeconv<128, 256, 256><<<1024, 256, edge_sm(128, 256), stream>>>(
      cat16, 512, 128, idxb, W4h, s4, h4, cat16, 256);

  // feat = lrelu(bn5(cat @ W5^T)) -> f16
  k_gemm_feat<<<dim3(256, 8), 256, 0, stream>>>(cat16, W5h, s5, h5, fe16);
  // fused heads -> d_out (f32, four regions)
  k_gemm_heads<<<dim3(256, 12), 256, 0, stream>>>(fe16, Wcat, cscale, cshift, cbias,
                                                  (float*)d_out);
  // sem = out3 @ Wsem^T + bsem
  k_sem<<<2048, 256, 0, stream>>>((const float*)d_out + 10485760, Wsem, bsem,
                                  (float*)d_out + 12582912);
}
